// ContrastiveEncoderShallowME_28097676050924
// MI455X (gfx1250) — matrix operand order repaired and checked by reading, not compiled, not run on hardware
//
#include <hip/hip_runtime.h>
#include <hip/hip_bf16.h>

typedef __attribute__((ext_vector_type(16))) _Float16 v16h;
typedef __attribute__((ext_vector_type(8)))  _Float16 v8h;
typedef __attribute__((ext_vector_type(8)))  float    v8f;

#define WMMA_F16(a, b, c) \
  __builtin_amdgcn_wmma_f32_16x16x32_f16(false, (a), false, (b), (short)0, (c), false, false)

// ---------------------------------------------------------------------------
// Conv weights (3,3,Cin,Cout) HWIO -> f16 [Cout][Kpad], K=(ky*3+kx)*Cin+ci,
// zero-padded K rows (flat HWIO index = k*Cout + n).
// ---------------------------------------------------------------------------
__global__ void cvt_conv_w(const float* __restrict__ w, _Float16* __restrict__ wt,
                           int K, int Kpad, int Cout) {
  int i = blockIdx.x * blockDim.x + threadIdx.x;
  int total = Cout * Kpad;
  if (i >= total) return;
  int n = i / Kpad;
  int k = i - n * Kpad;
  wt[i] = (k < K) ? (_Float16)w[k * Cout + n] : (_Float16)0.f;
}

// FC2/FC3 weights: lin_W is [DOUT][DIN] row-major == exactly the B layout.
__global__ void cvt_flat(const float* __restrict__ w, _Float16* __restrict__ wt, int total) {
  int i = blockIdx.x * blockDim.x + threadIdx.x;
  if (i < total) wt[i] = (_Float16)w[i];
}

// FC1 weights: activations are NHWC-flattened (k = (h*4+w)*256 + c), but the
// reference flattens CHW (k_ref = c*32 + h*4 + w). Permute during convert.
__global__ void cvt_fc1(const float* __restrict__ w, _Float16* __restrict__ wt) {
  int i = blockIdx.x * blockDim.x + threadIdx.x;
  if (i >= 512 * 8192) return;
  int n  = i >> 13;          // / 8192
  int kn = i & 8191;         // NHWC k : p*256 + c
  int p  = kn >> 8;
  int c  = kn & 255;
  wt[i] = (_Float16)w[n * 8192 + c * 32 + p];
}

// ---------------------------------------------------------------------------
// Implicit-GEMM 3x3 conv (pad=1) + optional BN + ReLU.
// Activations NHWC f16. Block = 128 threads = 4 waves; each wave owns MT
// consecutive 16-row M tiles x all Cout columns (MT sized so B fragments are
// reused MT times from registers and independent WMMAs cover RAW hazards).
// B chunk (32 x Cout) staged in LDS per K-step by the whole block.
// ---------------------------------------------------------------------------
template <typename TIN, int CIN, int COUT, int STRIDE, bool BN, int HIN, int WIN, int MT>
__global__ __launch_bounds__(128) void conv_k(
    const TIN* __restrict__ in, const _Float16* __restrict__ wt,
    _Float16* __restrict__ out,
    const float* __restrict__ bns, const float* __restrict__ bnb,
    const float* __restrict__ bnm, const float* __restrict__ bnv) {
  constexpr int HOUT = (STRIDE == 2) ? HIN / 2 : HIN;
  constexpr int WOUT = (STRIDE == 2) ? WIN / 2 : WIN;
  constexpr int K    = CIN * 9;
  constexpr int KPAD = ((K + 31) / 32) * 32;
  constexpr int NT   = COUT / 16;
  constexpr int NPIX = HOUT * WOUT;

  __shared__ __align__(64) _Float16 lb[32 * COUT];  // [n][kk] : lb[n*32+kk]

  const int tid   = threadIdx.x;
  const int wave  = tid >> 5;
  const int lane  = tid & 31;
  const int mbase = (blockIdx.x * 4 + wave) * (16 * MT);

  v8f acc[MT][NT];
#pragma unroll
  for (int mt = 0; mt < MT; ++mt)
#pragma unroll
    for (int t = 0; t < NT; ++t)
#pragma unroll
      for (int e = 0; e < 8; ++e) acc[mt][t][e] = 0.f;

  // A-fragment rows owned by this lane (16-bit A 16x32 layout, wave32)
  int iy0[MT], ix0[MT];
  const TIN* inp[MT];
#pragma unroll
  for (int mt = 0; mt < MT; ++mt) {
    const int row = mbase + mt * 16 + (lane & 15);
    const int img = row / NPIX;
    const int pix = row - img * NPIX;
    const int oy  = pix / WOUT;
    const int ox  = pix - oy * WOUT;
    iy0[mt] = oy * STRIDE - 1;
    ix0[mt] = ox * STRIDE - 1;
    inp[mt] = in + (size_t)img * CIN * HIN * WIN;  // NHWC per-image base
  }
  const int kA = (lane >> 4) * 8;   // A: half-wave K base
  const int kB = (lane >> 4) * 16;  // unused: B loads use kA (8*half), same k order as A

  for (int k0 = 0; k0 < K; k0 += 32) {
    __syncthreads();
    // cooperative stage of B chunk: lb[n*32+kk] = wt[n*KPAD + k0 + kk]
#pragma unroll 1
    for (int c = tid; c < COUT * 4; c += 128) {
      int n = c >> 2;
      int p = (c & 3) * 8;
      *(v8h*)(&lb[n * 32 + p]) = *(const v8h*)(wt + (size_t)n * KPAD + k0 + p);
    }
    __builtin_prefetch(wt + k0 + 32, 0, 0);  // global_prefetch_b8 next chunk
    __syncthreads();

    v16h a[MT];
#pragma unroll
    for (int mt = 0; mt < MT; ++mt) {
      if constexpr (CIN == 1) {
        // layer-0 path (NCHW==NHWC for C=1): scalar gather, single K-step
#pragma unroll
        for (int e = 0; e < 16; ++e) {
          int k = k0 + kA + (e & 7) + ((e >> 3) * 16);
          float v = 0.f;
          if (k < K) {
            int ky = k / 3;
            int kx = k - ky * 3;
            int iy = iy0[mt] + ky;
            int ix = ix0[mt] + kx;
            if ((unsigned)iy < (unsigned)HIN && (unsigned)ix < (unsigned)WIN)
              v = (float)inp[mt][(size_t)iy * WIN + ix];
          }
          a[mt][e] = (_Float16)v;
        }
      } else {
        // NHWC fast path: each 8-element half-group = one spatial position,
        // 8 contiguous channels -> single 16B load (ci0 multiple of 8, 8|CIN).
        v8h g0 = {}, g1 = {};
        {
          int kh = k0 + kA;
          int p  = kh / CIN;            // CIN pow2 -> shift
          int ci = kh & (CIN - 1);
          int ky = p / 3, kx = p - ky * 3;
          int iy = iy0[mt] + ky, ix = ix0[mt] + kx;
          if ((KPAD == K || p < 9) &&
              (unsigned)iy < (unsigned)HIN && (unsigned)ix < (unsigned)WIN)
            g0 = *(const v8h*)((const _Float16*)inp[mt] + (size_t)(iy * WIN + ix) * CIN + ci);
        }
        {
          int kh = k0 + kA + 16;
          int p  = kh / CIN;
          int ci = kh & (CIN - 1);
          int ky = p / 3, kx = p - ky * 3;
          int iy = iy0[mt] + ky, ix = ix0[mt] + kx;
          if ((KPAD == K || p < 9) &&
              (unsigned)iy < (unsigned)HIN && (unsigned)ix < (unsigned)WIN)
            g1 = *(const v8h*)((const _Float16*)inp[mt] + (size_t)(iy * WIN + ix) * CIN + ci);
        }
#pragma unroll
        for (int e = 0; e < 8; ++e) { a[mt][e] = g0[e]; a[mt][8 + e] = g1[e]; }
      }
    }

#pragma unroll
    for (int t = 0; t < NT; ++t) {
      v8h b0 = *(const v8h*)(&lb[(t * 16 + (lane & 15)) * 32 + kA]);
      v8h b1 = *(const v8h*)(&lb[(t * 16 + (lane & 15)) * 32 + kA + 16]);
      v16h b = __builtin_shufflevector(b0, b1, 0, 1, 2, 3, 4, 5, 6, 7, 8, 9, 10, 11, 12, 13, 14, 15);
#pragma unroll
      for (int mt = 0; mt < MT; ++mt)      // B reused from registers MT times
        acc[mt][t] = WMMA_F16(a[mt], b, acc[mt][t]);
    }
  }

  // Epilogue: C/D layout -> M = v + 8*(lane>>4), N = lane&15 (per tile).
  // NHWC store: flat pixel index == GEMM row -> addr = rr*COUT + c (no divs).
#pragma unroll
  for (int mt = 0; mt < MT; ++mt) {
    const int rb = mbase + mt * 16 + ((lane >> 4) * 8);
#pragma unroll
    for (int t = 0; t < NT; ++t) {
      const int c = t * 16 + (lane & 15);
      float inv = 1.f, beta = 0.f;
      if (BN) {
        inv  = bns[c] * rsqrtf(bnv[c] + 1e-5f);
        beta = bnb[c] - bnm[c] * inv;
      }
#pragma unroll
      for (int v = 0; v < 8; ++v) {
        float val = acc[mt][t][v];
        if (BN) val = val * inv + beta;
        val = fmaxf(val, 0.f);
        out[(size_t)(rb + v) * COUT + c] = (_Float16)val;
      }
    }
  }
}

// ---------------------------------------------------------------------------
// FC GEMM: out[m,n] = act( sum_k in[m,k] * W[n,k] + bias[n] )
// One wave per block; 16(M) x 64(N) tile. ACT: 0=SiLU, 1=tanh.
// ---------------------------------------------------------------------------
template <int DIN, int DOUT, int ACT, typename TOUT>
__global__ __launch_bounds__(32) void fc_k(
    const _Float16* __restrict__ in, const _Float16* __restrict__ wt,
    const float* __restrict__ bias, TOUT* __restrict__ out) {
  const int lane  = threadIdx.x & 31;
  const int mbase = blockIdx.x * 16;
  const int nbase = blockIdx.y * 64;

  v8f acc[4];
#pragma unroll
  for (int t = 0; t < 4; ++t)
#pragma unroll
    for (int e = 0; e < 8; ++e) acc[t][e] = 0.f;

  const _Float16* arow = in + (size_t)(mbase + (lane & 15)) * DIN;
  const int kA = (lane >> 4) * 8;
  const int kB = (lane >> 4) * 16;

  for (int k0 = 0; k0 < DIN; k0 += 32) {
    v8h a0 = *(const v8h*)(arow + k0 + kA);
    v8h a1 = *(const v8h*)(arow + k0 + kA + 16);
    v16h a;
#pragma unroll
    for (int e = 0; e < 8; ++e) { a[e] = a0[e]; a[8 + e] = a1[e]; }
#pragma unroll
    for (int t = 0; t < 4; ++t) {
      v8h b0 = *(const v8h*)(wt + (size_t)(nbase + t * 16 + (lane & 15)) * DIN + k0 + kA);
      v8h b1 = *(const v8h*)(wt + (size_t)(nbase + t * 16 + (lane & 15)) * DIN + k0 + kA + 16);
      v16h b = __builtin_shufflevector(b0, b1, 0, 1, 2, 3, 4, 5, 6, 7, 8, 9, 10, 11, 12, 13, 14, 15);
      acc[t] = WMMA_F16(a, b, acc[t]);
    }
  }

  const int rb = mbase + ((lane >> 4) * 8);
#pragma unroll
  for (int t = 0; t < 4; ++t) {
    int n = nbase + t * 16 + (lane & 15);
    float bv = bias[n];
#pragma unroll
    for (int v = 0; v < 8; ++v) {
      float val = acc[t][v] + bv;
      if (ACT == 0) val = val / (1.f + __expf(-val));   // SiLU
      else          val = tanhf(val);
      out[(size_t)(rb + v) * DOUT + n] = (TOUT)val;
    }
  }
}

// ---------------------------------------------------------------------------
extern "C" void kernel_launch(void* const* d_in, const int* in_sizes, int n_in,
                              void* d_out, int out_size, void* d_ws, size_t ws_size,
                              hipStream_t stream) {
  (void)in_sizes; (void)n_in; (void)out_size; (void)ws_size;

  // inputs: x, batch_size, conv_kernels[10], bn_params[6][4], lin_W[3], lin_b[3]
  const float* x = (const float*)d_in[0];
  const float* cw[10];
  for (int i = 0; i < 10; ++i) cw[i] = (const float*)d_in[2 + i];
  const float* bnp[6][4];
  for (int j = 0; j < 6; ++j)
    for (int q = 0; q < 4; ++q) bnp[j][q] = (const float*)d_in[12 + j * 4 + q];
  const float* lw[3];
  const float* lbv[3];
  for (int i = 0; i < 3; ++i) { lw[i] = (const float*)d_in[36 + i]; lbv[i] = (const float*)d_in[39 + i]; }

  // workspace layout
  char* ws = (char*)d_ws;
  size_t off = 0;
  auto alloc = [&](size_t bytes) -> void* {
    void* p = ws + off;
    off = (off + bytes + 255) & ~(size_t)255;
    return p;
  };
  const size_t ACT_MAX = (size_t)128 * 16 * 128 * 64;  // largest activation (elems)
  _Float16* actA = (_Float16*)alloc(ACT_MAX * 2);
  _Float16* actB = (_Float16*)alloc(ACT_MAX * 2);

  static const int KP[10] = {32, 160, 160, 288, 288, 576, 576, 1152, 1152, 2304};
  static const int KK[10] = {9, 144, 144, 288, 288, 576, 576, 1152, 1152, 2304};
  static const int CO[10] = {16, 16, 32, 32, 64, 64, 128, 128, 256, 256};
  _Float16* wc[10];
  for (int i = 0; i < 10; ++i) wc[i] = (_Float16*)alloc((size_t)KP[i] * CO[i] * 2);
  static const int FD[3][2] = {{8192, 512}, {512, 256}, {256, 128}};
  _Float16* wf[3];
  for (int i = 0; i < 3; ++i) wf[i] = (_Float16*)alloc((size_t)FD[i][0] * FD[i][1] * 2);

  // convert weights to f16 (packed layouts)
  for (int i = 0; i < 10; ++i) {
    int total = CO[i] * KP[i];
    cvt_conv_w<<<(total + 255) / 256, 256, 0, stream>>>(cw[i], wc[i], KK[i], KP[i], CO[i]);
  }
  cvt_fc1<<<(512 * 8192 + 255) / 256, 256, 0, stream>>>(lw[0], wf[0]);
  for (int i = 1; i < 3; ++i) {
    int total = FD[i][0] * FD[i][1];
    cvt_flat<<<(total + 255) / 256, 256, 0, stream>>>(lw[i], wf[i], total);
  }

  const float* np = nullptr;
  // conv stack, activations NHWC f16. grid = B*Hout*Wout / (64*MT)
  conv_k<float,     1,  16, 2, false, 256, 128, 4><<< 4096, 128, 0, stream>>>(x,    wc[0], actA, np, np, np, np);
  conv_k<_Float16, 16,  16, 1, false, 128,  64, 4><<< 4096, 128, 0, stream>>>(actA, wc[1], actB, np, np, np, np);
  conv_k<_Float16, 16,  32, 2, false, 128,  64, 4><<< 1024, 128, 0, stream>>>(actB, wc[2], actA, np, np, np, np);
  conv_k<_Float16, 32,  32, 1, false,  64,  32, 4><<< 1024, 128, 0, stream>>>(actA, wc[3], actB, np, np, np, np);
  conv_k<_Float16, 32,  64, 2, true,   64,  32, 2><<<  512, 128, 0, stream>>>(actB, wc[4], actA, bnp[0][0], bnp[0][1], bnp[0][2], bnp[0][3]);
  conv_k<_Float16, 64,  64, 1, true,   32,  16, 2><<<  512, 128, 0, stream>>>(actA, wc[5], actB, bnp[1][0], bnp[1][1], bnp[1][2], bnp[1][3]);
  conv_k<_Float16, 64, 128, 2, true,   32,  16, 2><<<  128, 128, 0, stream>>>(actB, wc[6], actA, bnp[2][0], bnp[2][1], bnp[2][2], bnp[2][3]);
  conv_k<_Float16,128, 128, 1, true,   16,   8, 2><<<  128, 128, 0, stream>>>(actA, wc[7], actB, bnp[3][0], bnp[3][1], bnp[3][2], bnp[3][3]);
  conv_k<_Float16,128, 256, 2, true,   16,   8, 1><<<   64, 128, 0, stream>>>(actB, wc[8], actA, bnp[4][0], bnp[4][1], bnp[4][2], bnp[4][3]);
  conv_k<_Float16,256, 256, 1, true,    8,   4, 1><<<   64, 128, 0, stream>>>(actA, wc[9], actB, bnp[5][0], bnp[5][1], bnp[5][2], bnp[5][3]);

  // FC head: [128,8192] -> 512 (SiLU) -> 256 (SiLU) -> 128 (tanh, f32 out)
  fc_k<8192, 512, 0, _Float16><<<dim3(8, 8), 32, 0, stream>>>(actB, wf[0], lbv[0], actA);
  fc_k< 512, 256, 0, _Float16><<<dim3(8, 4), 32, 0, stream>>>(actA, wf[1], lbv[1], actB);
  fc_k< 256, 128, 1, float   ><<<dim3(8, 2), 32, 0, stream>>>(actB, wf[2], lbv[2], (float*)d_out);
}
